// Attention_55860344652376
// MI455X (gfx1250) — compile-verified, matrix operand order checked by reading, not run on hardware
//
#include <hip/hip_runtime.h>
#include <hip/hip_bf16.h>
#include <stdint.h>

// CDNA5 / gfx1250 wave32 WMMA attention pipeline.
// d_out layout (floats): [ z (S*E) | scores (S*S) | Q | V | K ], S=8192, E=1024.
//
// Fast path (needs ~198MB workspace): all GEMM operands are f16 K-major in
// d_ws, streamed into LDS with GLOBAL_LOAD_ASYNC_TO_LDS_B128 (double
// buffered, s_wait_asynccnt), matrix math via v_wmma_f32_16x16x32_f16 with
// f32 accumulation. Projection GEMMs write fp32 results to d_out AND f16
// copies to d_ws; softmax writes normalized f16 scores for the PV GEMM.
// Fallback path (small ws): round-1 sync-staging fp32->f16 pipeline.

typedef __attribute__((ext_vector_type(16))) _Float16 v16h;
typedef __attribute__((ext_vector_type(8)))  _Float16 v8h;
typedef __attribute__((ext_vector_type(4)))  _Float16 v4h;
typedef __attribute__((ext_vector_type(8)))  float    v8f;

#define BM 128
#define BN 128
#define BK 32
#define BKP 40   // padded LDS row stride (halfwords): 80B rows -> 16B-aligned
                 // ds_load_b128 fragments, 20-bank stride -> conflict-free.

// ===========================================================================
// FAST PATH: async f16 GEMM. A [M,K] row-major f16, B [N,K] row-major f16.
// 256 thr = 8 waves (2x4 wave grid), 64x32 per wave = 4x2 WMMA tiles.
// OUTF16: 0 = fp32 only; 1 = also F[gm*ldf+gn] f16; 2 = also F[gn*ldf+gm].
// ===========================================================================
template <bool CAUSAL_OUT, bool CAUSAL_K, int OUTF16>
__global__ __launch_bounds__(256) void gemm_async_f16(
    const _Float16* __restrict__ A, const _Float16* __restrict__ B,
    float* __restrict__ C, _Float16* __restrict__ F,
    int M, int N, int K, int lda, int ldb, int ldc, int ldf, float scale) {
  const int blockN = blockIdx.x * BN;
  const int blockM = blockIdx.y * BM;
  const int tid    = threadIdx.x;

  if (CAUSAL_OUT) {
    if (blockN > blockM + (BM - 1)) {   // tile strictly above diagonal
      const float ninf = -__builtin_inff();
#pragma unroll 4
      for (int i = 0; i < (BM * BN) / 256; ++i) {
        int idx = tid + i * 256;
        C[(size_t)(blockM + (idx >> 7)) * ldc + (blockN + (idx & (BN - 1)))] = ninf;
      }
      return;
    }
  }

  __shared__ _Float16 sA[2][BM * BKP];
  __shared__ _Float16 sB[2][BN * BKP];

  const int lane = tid & 31;
  const int wave = tid >> 5;
  const int wr   = wave >> 2;
  const int wc   = wave & 3;

  // Async staging: wave w stages tile rows [w*16, w*16+16). One
  // global_load_async_to_lds_b128 moves 32 lanes x 16B = 8 rows x 64B.
  const int srow0 = wave * 16 + (lane >> 2);
  const int srow1 = srow0 + 8;
  const int sseg  = (lane & 3) * 16;                 // 16B segment in 64B row
  const int voffA0 = srow0 * lda * 2 + sseg;         // global byte offsets
  const int voffA1 = srow1 * lda * 2 + sseg;
  const int voffB0 = srow0 * ldb * 2 + sseg;
  const int voffB1 = srow1 * ldb * 2 + sseg;
  const unsigned ldsA  = (unsigned)(uintptr_t)&sA[0][0];  // LDS byte offsets
  const unsigned ldsB  = (unsigned)(uintptr_t)&sB[0][0];
  const unsigned loff0 = srow0 * (BKP * 2) + sseg;
  const unsigned loff1 = srow1 * (BKP * 2) + sseg;
  const unsigned bufBytes = (unsigned)(BM * BKP * sizeof(_Float16));

  auto issue_tile = [&](int buf, int k0) {
    const _Float16* aB = A + (size_t)blockM * lda + k0;
    const _Float16* bB = B + (size_t)blockN * ldb + k0;
    const unsigned la0 = ldsA + buf * bufBytes + loff0;
    const unsigned la1 = ldsA + buf * bufBytes + loff1;
    const unsigned lb0 = ldsB + buf * bufBytes + loff0;
    const unsigned lb1 = ldsB + buf * bufBytes + loff1;
    asm volatile("global_load_async_to_lds_b128 %0, %1, %4\n\t"
                 "global_load_async_to_lds_b128 %2, %3, %4"
                 :: "v"(la0), "v"(voffA0), "v"(la1), "v"(voffA1), "s"(aB)
                 : "memory");
    asm volatile("global_load_async_to_lds_b128 %0, %1, %4\n\t"
                 "global_load_async_to_lds_b128 %2, %3, %4"
                 :: "v"(lb0), "v"(voffB0), "v"(lb1), "v"(voffB1), "s"(bB)
                 : "memory");
  };

  v8f acc[4][2];
#pragma unroll
  for (int mi = 0; mi < 4; ++mi)
#pragma unroll
    for (int ni = 0; ni < 2; ++ni)
      acc[mi][ni] = (v8f){0.f, 0.f, 0.f, 0.f, 0.f, 0.f, 0.f, 0.f};

  // Fragment addressing (ISA 7.12.2, 16-bit 16x32 operands).
  const int frow = lane & 15;
  const int fseg = (lane >> 4) * 8;

  const int kEnd = CAUSAL_K ? ((blockM + BM < K) ? blockM + BM : K) : K;

  issue_tile(0, 0);
  int buf = 0;
  for (int k0 = 0; k0 < kEnd; k0 += BK) {
    if (k0 + BK < kEnd) {
      issue_tile(buf ^ 1, k0 + BK);
      asm volatile("s_wait_asynccnt 0x4" ::: "memory");  // current tile done
    } else {
      asm volatile("s_wait_asynccnt 0x0" ::: "memory");
    }
    __syncthreads();   // all waves' async stores to LDS visible

    v16h af[4], bfr[2];
#pragma unroll
    for (int mi = 0; mi < 4; ++mi) {
      const _Float16* p = &sA[buf][(wr * 64 + mi * 16 + frow) * BKP + fseg];
      v8h lo = *(const v8h*)p;
      v8h hi = *(const v8h*)(p + 16);
      af[mi] = __builtin_shufflevector(lo, hi, 0, 1, 2, 3, 4, 5, 6, 7,
                                       8, 9, 10, 11, 12, 13, 14, 15);
    }
#pragma unroll
    for (int ni = 0; ni < 2; ++ni) {
      const _Float16* p = &sB[buf][(wc * 32 + ni * 16 + frow) * BKP + fseg];
      v8h lo = *(const v8h*)p;
      v8h hi = *(const v8h*)(p + 16);
      bfr[ni] = __builtin_shufflevector(lo, hi, 0, 1, 2, 3, 4, 5, 6, 7,
                                        8, 9, 10, 11, 12, 13, 14, 15);
    }
#pragma unroll
    for (int mi = 0; mi < 4; ++mi)
#pragma unroll
      for (int ni = 0; ni < 2; ++ni)
        acc[mi][ni] = __builtin_amdgcn_wmma_f32_16x16x32_f16(
            false, af[mi], false, bfr[ni], (short)0, acc[mi][ni], false, false);
    __syncthreads();   // done reading this buffer
    buf ^= 1;
  }

  const int nIdx    = lane & 15;
  const int rowHalf = (lane >> 4) * 8;
  const float ninf  = -__builtin_inff();
#pragma unroll
  for (int mi = 0; mi < 4; ++mi) {
#pragma unroll
    for (int ni = 0; ni < 2; ++ni) {
      const int gm0 = blockM + wr * 64 + mi * 16 + rowHalf;
      const int gn  = blockN + wc * 32 + ni * 16 + nIdx;
#pragma unroll
      for (int r = 0; r < 8; ++r) {
        const int gm = gm0 + r;
        float vv = acc[mi][ni][r] * scale;
        if (CAUSAL_OUT && gn > gm) vv = ninf;
        C[(size_t)gm * ldc + gn] = vv;
        if (OUTF16 == 1) F[(size_t)gm * ldf + gn] = (_Float16)vv;
        if (OUTF16 == 2) F[(size_t)gn * ldf + gm] = (_Float16)vv;
      }
    }
  }
}

// ===========================================================================
// FALLBACK: round-1 sync-staging GEMM, fp32 operands converted on the fly.
// ===========================================================================
template <bool TRANSB, bool CAUSAL_OUT, bool CAUSAL_K>
__global__ __launch_bounds__(256) void gemm_wmma_f16(
    const float* __restrict__ A, const float* __restrict__ B,
    float* __restrict__ C, int M, int N, int K,
    int lda, int ldb, int ldc, float scale) {
  const int blockN = blockIdx.x * BN;
  const int blockM = blockIdx.y * BM;
  const int tid    = threadIdx.x;

  if (CAUSAL_OUT) {
    if (blockN > blockM + (BM - 1)) {
      const float ninf = -__builtin_inff();
#pragma unroll 4
      for (int i = 0; i < (BM * BN) / 256; ++i) {
        int idx = tid + i * 256;
        C[(size_t)(blockM + (idx >> 7)) * ldc + (blockN + (idx & (BN - 1)))] = ninf;
      }
      return;
    }
  }

  __shared__ _Float16 sA[BM * BKP];
  __shared__ _Float16 sB[BN * BKP];

  const int lane = tid & 31;
  const int wave = tid >> 5;
  const int wr   = wave >> 2;
  const int wc   = wave & 3;

  v8f acc[4][2];
#pragma unroll
  for (int mi = 0; mi < 4; ++mi)
#pragma unroll
    for (int ni = 0; ni < 2; ++ni)
      acc[mi][ni] = (v8f){0.f, 0.f, 0.f, 0.f, 0.f, 0.f, 0.f, 0.f};

  const int arow = tid >> 1;
  const int acol = (tid & 1) * 16;
  const float* Abase = A + (size_t)(blockM + arow) * lda + acol;

  const int frow = lane & 15;
  const int fseg = (lane >> 4) * 8;

  const int kEnd = CAUSAL_K ? ((blockM + BM < K) ? blockM + BM : K) : K;

  for (int k0 = 0; k0 < kEnd; k0 += BK) {
    {
      const float4 a0 = *(const float4*)(Abase + k0 + 0);
      const float4 a1 = *(const float4*)(Abase + k0 + 4);
      const float4 a2 = *(const float4*)(Abase + k0 + 8);
      const float4 a3 = *(const float4*)(Abase + k0 + 12);
      v8h p0 = {(_Float16)a0.x, (_Float16)a0.y, (_Float16)a0.z, (_Float16)a0.w,
                (_Float16)a1.x, (_Float16)a1.y, (_Float16)a1.z, (_Float16)a1.w};
      v8h p1 = {(_Float16)a2.x, (_Float16)a2.y, (_Float16)a2.z, (_Float16)a2.w,
                (_Float16)a3.x, (_Float16)a3.y, (_Float16)a3.z, (_Float16)a3.w};
      *(v8h*)(&sA[arow * BKP + acol])     = p0;
      *(v8h*)(&sA[arow * BKP + acol + 8]) = p1;
    }
    if (TRANSB) {
      const float* Bbase = B + (size_t)(blockN + arow) * ldb + acol + k0;
      const float4 b0 = *(const float4*)(Bbase + 0);
      const float4 b1 = *(const float4*)(Bbase + 4);
      const float4 b2 = *(const float4*)(Bbase + 8);
      const float4 b3 = *(const float4*)(Bbase + 12);
      v8h p0 = {(_Float16)b0.x, (_Float16)b0.y, (_Float16)b0.z, (_Float16)b0.w,
                (_Float16)b1.x, (_Float16)b1.y, (_Float16)b1.z, (_Float16)b1.w};
      v8h p1 = {(_Float16)b2.x, (_Float16)b2.y, (_Float16)b2.z, (_Float16)b2.w,
                (_Float16)b3.x, (_Float16)b3.y, (_Float16)b3.z, (_Float16)b3.w};
      *(v8h*)(&sB[arow * BKP + acol])     = p0;
      *(v8h*)(&sB[arow * BKP + acol + 8]) = p1;
    } else {
      const int bk   = tid >> 3;
      const int nseg = (tid & 7) * 16;
      const float* Bp = B + (size_t)(k0 + bk) * ldb + blockN + nseg;
      const float4 b0 = *(const float4*)(Bp + 0);
      const float4 b1 = *(const float4*)(Bp + 4);
      const float4 b2 = *(const float4*)(Bp + 8);
      const float4 b3 = *(const float4*)(Bp + 12);
      float v[16] = {b0.x, b0.y, b0.z, b0.w, b1.x, b1.y, b1.z, b1.w,
                     b2.x, b2.y, b2.z, b2.w, b3.x, b3.y, b3.z, b3.w};
#pragma unroll
      for (int e = 0; e < 16; ++e)
        sB[(nseg + e) * BKP + bk] = (_Float16)v[e];
    }
    __syncthreads();

    v16h af[4], bfr[2];
#pragma unroll
    for (int mi = 0; mi < 4; ++mi) {
      const _Float16* p = &sA[(wr * 64 + mi * 16 + frow) * BKP + fseg];
      v8h lo = *(const v8h*)p;
      v8h hi = *(const v8h*)(p + 16);
      af[mi] = __builtin_shufflevector(lo, hi, 0, 1, 2, 3, 4, 5, 6, 7,
                                       8, 9, 10, 11, 12, 13, 14, 15);
    }
#pragma unroll
    for (int ni = 0; ni < 2; ++ni) {
      const _Float16* p = &sB[(wc * 32 + ni * 16 + frow) * BKP + fseg];
      v8h lo = *(const v8h*)p;
      v8h hi = *(const v8h*)(p + 16);
      bfr[ni] = __builtin_shufflevector(lo, hi, 0, 1, 2, 3, 4, 5, 6, 7,
                                        8, 9, 10, 11, 12, 13, 14, 15);
    }
#pragma unroll
    for (int mi = 0; mi < 4; ++mi)
#pragma unroll
      for (int ni = 0; ni < 2; ++ni)
        acc[mi][ni] = __builtin_amdgcn_wmma_f32_16x16x32_f16(
            false, af[mi], false, bfr[ni], (short)0, acc[mi][ni], false, false);
    __syncthreads();
  }

  const int nIdx    = lane & 15;
  const int rowHalf = (lane >> 4) * 8;
  const float ninf  = -__builtin_inff();
#pragma unroll
  for (int mi = 0; mi < 4; ++mi) {
#pragma unroll
    for (int ni = 0; ni < 2; ++ni) {
      const int gm0 = blockM + wr * 64 + mi * 16 + rowHalf;
      const int gn  = blockN + wc * 32 + ni * 16 + nIdx;
#pragma unroll
      for (int r = 0; r < 8; ++r) {
        const int gm = gm0 + r;
        float vv = acc[mi][ni][r] * scale;
        if (CAUSAL_OUT && gn > gm) vv = ninf;
        C[(size_t)gm * ldc + gn] = vv;
      }
    }
  }
}

// ---------------------------------------------------------------------------
// Helpers: fp32 -> f16 convert; fp32 [D,E] -> f16 transposed [E,D].
// ---------------------------------------------------------------------------
__global__ __launch_bounds__(256) void cvt_f32_f16_vec(
    const float* __restrict__ in, _Float16* __restrict__ out, int n4) {
  const int i = blockIdx.x * 256 + threadIdx.x;
  if (i < n4) {
    const float4 v = ((const float4*)in)[i];
    v4h h = {(_Float16)v.x, (_Float16)v.y, (_Float16)v.z, (_Float16)v.w};
    ((v4h*)out)[i] = h;
  }
}

__global__ __launch_bounds__(256) void transpose_cvt_f16(
    const float* __restrict__ in, _Float16* __restrict__ out, int D, int E) {
  __shared__ float t[32][33];
  const int bx = blockIdx.x * 32;   // E
  const int by = blockIdx.y * 32;   // D
  const int x = threadIdx.x, y = threadIdx.y;   // 32 x 8
#pragma unroll
  for (int i = 0; i < 32; i += 8)
    t[y + i][x] = in[(size_t)(by + y + i) * E + bx + x];
  __syncthreads();
#pragma unroll
  for (int i = 0; i < 32; i += 8)
    out[(size_t)(bx + y + i) * D + by + x] = (_Float16)t[x][y + i];
}

// ---------------------------------------------------------------------------
// In-place row softmax; optionally emits normalized f16 scores.
// ---------------------------------------------------------------------------
__global__ __launch_bounds__(256) void softmax_rows(
    float* __restrict__ scores, int S, _Float16* __restrict__ f16out) {
  const size_t row = blockIdx.x;
  float* p = scores + row * S;
  __shared__ float red[256];
  const int tid = threadIdx.x;

  float m = -__builtin_inff();
  for (int c = tid; c < S; c += 256) m = fmaxf(m, p[c]);
  red[tid] = m;
  __syncthreads();
  for (int s = 128; s > 0; s >>= 1) {
    if (tid < s) red[tid] = fmaxf(red[tid], red[tid + s]);
    __syncthreads();
  }
  m = red[0];
  __syncthreads();

  float sum = 0.f;
  for (int c = tid; c < S; c += 256) {
    float e = __expf(p[c] - m);
    p[c] = e;
    sum += e;
  }
  red[tid] = sum;
  __syncthreads();
  for (int s = 128; s > 0; s >>= 1) {
    if (tid < s) red[tid] += red[tid + s];
    __syncthreads();
  }
  const float inv = 1.0f / red[0];
  if (f16out) {
    _Float16* q = f16out + row * S;
    for (int c = tid; c < S; c += 256) {
      const float s = p[c] * inv;
      p[c] = s;
      q[c] = (_Float16)s;
    }
  } else {
    for (int c = tid; c < S; c += 256) p[c] *= inv;
  }
}

extern "C" void kernel_launch(void* const* d_in, const int* in_sizes, int n_in,
                              void* d_out, int out_size, void* d_ws,
                              size_t ws_size, hipStream_t stream) {
  (void)in_sizes; (void)n_in; (void)out_size;
  const float* x  = (const float*)d_in[0];
  const float* wq = (const float*)d_in[1];
  const float* wk = (const float*)d_in[2];
  const float* wv = (const float*)d_in[3];

  const int S = 8192, D = 1024, E = 1024;
  const size_t SD = (size_t)S * D, SE = (size_t)S * E;
  const size_t SS = (size_t)S * S, DE = (size_t)D * E;

  float* out    = (float*)d_out;
  float* z      = out;                     // [S,E]
  float* scores = z + SE;                  // [S,S]
  float* Q      = scores + SS;             // [S,E]
  float* V      = Q + SE;                  // [S,E]
  float* Km     = V + SE;                  // [S,E]

  const dim3 blk(256);
  const dim3 gProj(E / BN, S / BM);        // 8 x 64
  const dim3 gAttn(S / BN, S / BM);        // 64 x 64

  const size_t need = (SD + 3 * DE + 2 * SE + SE + SS) * sizeof(_Float16);
  if (ws_size >= need) {
    // ---- fast path: f16 operands, async-to-LDS GEMMs ----
    _Float16* xf  = (_Float16*)d_ws;       // [S,D]
    _Float16* wqT = xf + SD;               // [E,D]
    _Float16* wkT = wqT + DE;
    _Float16* wvT = wkT + DE;
    _Float16* Qf  = wvT + DE;              // [S,E]
    _Float16* Kf  = Qf + SE;               // [S,E]
    _Float16* Vt  = Kf + SE;               // [E,S]
    _Float16* Pf  = Vt + SE;               // [S,S]

    cvt_f32_f16_vec<<<dim3((unsigned)(SD / 4 / 256)), blk, 0, stream>>>(
        x, xf, (int)(SD / 4));
    const dim3 gT(E / 32, D / 32), bT(32, 8);
    transpose_cvt_f16<<<gT, bT, 0, stream>>>(wq, wqT, D, E);
    transpose_cvt_f16<<<gT, bT, 0, stream>>>(wk, wkT, D, E);
    transpose_cvt_f16<<<gT, bT, 0, stream>>>(wv, wvT, D, E);

    gemm_async_f16<false, false, 1><<<gProj, blk, 0, stream>>>(
        xf, wqT, Q, Qf, S, E, D, D, D, E, E, 1.0f);
    gemm_async_f16<false, false, 1><<<gProj, blk, 0, stream>>>(
        xf, wkT, Km, Kf, S, E, D, D, D, E, E, 1.0f);
    gemm_async_f16<false, false, 2><<<gProj, blk, 0, stream>>>(
        xf, wvT, V, Vt, S, E, D, D, D, E, S, 1.0f);

    gemm_async_f16<true, false, 0><<<gAttn, blk, 0, stream>>>(
        Qf, Kf, scores, (_Float16*)nullptr, S, S, E, E, E, S, 0, 1.0f / 32.0f);

    softmax_rows<<<dim3(S), blk, 0, stream>>>(scores, S, Pf);

    gemm_async_f16<false, true, 0><<<gProj, blk, 0, stream>>>(
        Pf, Vt, z, (_Float16*)nullptr, S, E, S, S, S, E, 0, 1.0f);
  } else {
    // ---- fallback: fp32 operands, sync staging ----
    gemm_wmma_f16<false, false, false><<<gProj, blk, 0, stream>>>(
        x, wq, Q, S, E, D, D, E, E, 1.0f);
    gemm_wmma_f16<false, false, false><<<gProj, blk, 0, stream>>>(
        x, wk, Km, S, E, D, D, E, E, 1.0f);
    gemm_wmma_f16<false, false, false><<<gProj, blk, 0, stream>>>(
        x, wv, V, S, E, D, D, E, E, 1.0f);
    gemm_wmma_f16<true, true, false><<<gAttn, blk, 0, stream>>>(
        Q, Km, scores, S, S, E, E, E, S, 1.0f / 32.0f);
    softmax_rows<<<dim3(S), blk, 0, stream>>>(scores, S, (_Float16*)nullptr);
    gemm_wmma_f16<false, false, true><<<gProj, blk, 0, stream>>>(
        scores, V, z, S, E, S, S, E, E, 1.0f);
  }
}
